// MambaBlock_12644383719444
// MI455X (gfx1250) — matrix operand order repaired and checked by reading, not compiled, not run on hardware
//
#include <hip/hip_runtime.h>
#include <hip/hip_bf16.h>
#include <math.h>

#define D_MODEL 256
#define EPS 1e-5f
#define ROWS 32               // rows per workgroup

typedef _Float16 h8  __attribute__((ext_vector_type(8)));
typedef _Float16 h16 __attribute__((ext_vector_type(16)));
typedef float    f8  __attribute__((ext_vector_type(8)));
typedef unsigned int u32x4 __attribute__((ext_vector_type(4)));
typedef int i32x4 __attribute__((ext_vector_type(4)));
typedef int i32x8 __attribute__((ext_vector_type(8)));

#define HAVE_TDM __has_builtin(__builtin_amdgcn_tensor_load_to_lds)

// f16 weight layout inside d_ws (element offsets, all 16B-aligned)
#define OFF_INPROJ  0          // [512][256]
#define OFF_XPROJ   131072     // [48][256]
#define OFF_DTPROJ  143360     // [256][32]  (K padded 16 -> 32 with zeros)
#define OFF_OUTPROJ 151552     // [256][256]
#define OFF_PROJ    217088     // [256][256]
#define WS_HALFS    282624

// ---------------------------------------------------------------- prep: f32 -> f16 weights
__global__ void prep_weights(const float* __restrict__ inproj,
                             const float* __restrict__ xproj,
                             const float* __restrict__ dtproj,
                             const float* __restrict__ outproj,
                             const float* __restrict__ proj,
                             _Float16* __restrict__ wh) {
  int i = blockIdx.x * blockDim.x + threadIdx.x;
  if (i < OFF_XPROJ) {
    wh[i] = (_Float16)inproj[i];
  } else if (i < OFF_DTPROJ) {
    wh[i] = (_Float16)xproj[i - OFF_XPROJ];
  } else if (i < OFF_OUTPROJ) {
    int j = i - OFF_DTPROJ; int n = j >> 5, k = j & 31;
    wh[i] = (k < 16) ? (_Float16)dtproj[n * 16 + k] : (_Float16)0.f;
  } else if (i < OFF_PROJ) {
    wh[i] = (_Float16)outproj[i - OFF_OUTPROJ];
  } else if (i < WS_HALFS) {
    wh[i] = (_Float16)proj[i - OFF_PROJ];
  }
}

// ---------------------------------------------------------------- WMMA fragment loaders
// A 16x32 f16 (row-major source, leading dim ld): lane m=l&15;
//   lanes 0-15 hold K 0..7 & 16..23, lanes 16-31 hold K 8..15 & 24..31 (ISA 7.12.2)
__device__ __forceinline__ h16 ld_a(const _Float16* p0, int ld, int lane) {
  const _Float16* p = p0 + (lane & 15) * ld + ((lane >> 4) ? 8 : 0);
  h8 lo = *(const h8*)p;
  h8 hi = *(const h8*)(p + 16);
  h16 r;
#pragma unroll
  for (int i = 0; i < 8; ++i) { r[i] = lo[i]; r[i + 8] = hi[i]; }
  return r;
}
// B 32x16 f16 from W[N][K] row-major: B(k,n)=W[n][k]; lane n=l&15;
//   lanes 0-15 hold K 0..7 & 16..23, lanes 16-31 hold K 8..15 & 24..31 (matches ld_a)
__device__ __forceinline__ h16 ld_b(const _Float16* W, int K, int n0, int k0, int lane) {
  const _Float16* p = W + (size_t)(n0 + (lane & 15)) * K + k0 + ((lane >> 4) ? 8 : 0);
  h8 lo = *(const h8*)p;
  h8 hi = *(const h8*)(p + 16);
  h16 r;
#pragma unroll
  for (int i = 0; i < 8; ++i) { r[i] = lo[i]; r[i + 8] = hi[i]; }
  return r;
}

__device__ __forceinline__ f8 wmma_step(h16 a, h16 b, f8 c) {
  return __builtin_amdgcn_wmma_f32_16x16x32_f16(false, a, false, b, (short)0, c, false, false);
}

__device__ __forceinline__ float silu(float v) { return v / (1.f + __expf(-v)); }

// epilogue for GEMM1 tiles: xi -> xc (conv+SiLU) or z -> silu(z)
__device__ __forceinline__ void epi_xz(f8 acc, int n0, int m0, int l16, int hi8,
                                       const float* __restrict__ conv_w,
                                       const float* __restrict__ conv_b,
                                       _Float16* bufB, _Float16* zbuf) {
  const int c = n0 + l16;
  if (n0 < 256) {
    float cw1 = conv_w[2 * c + 1];
    float cb = conv_b[c];
#pragma unroll
    for (int r = 0; r < 8; ++r) {
      int row = m0 + r + hi8;
      bufB[row * 256 + c] = (_Float16)silu(acc[r] * cw1 + cb);
    }
  } else {
    int c2 = c - 256;
#pragma unroll
    for (int r = 0; r < 8; ++r) {
      int row = m0 + r + hi8;
      zbuf[row * 256 + c2] = (_Float16)silu(acc[r]);
    }
  }
}

// epilogue for dt GEMM: softplus + full elementwise y -> bufA (f16)
__device__ __forceinline__ void epi_y(f8 acc, int n0, int m0, int l16, int hi8,
                                      const float* __restrict__ dt_proj_b,
                                      const float* __restrict__ Dvec,
                                      const _Float16* bufB, const _Float16* zbuf,
                                      const float* bc_s, _Float16* bufA) {
  const int c = n0 + l16;
  float dtb = dt_proj_b[c];
  float Dv = Dvec[c];
#pragma unroll
  for (int r = 0; r < 8; ++r) {
    int row = m0 + r + hi8;
    float v = acc[r] + dtb;
    float dt = (v > 20.f) ? v : log1pf(__expf(v));
    float xc = (float)bufB[row * 256 + c];
    float sz = (float)zbuf[row * 256 + c];
    float y = (dt * bc_s[row] + Dv) * xc * sz;
    bufA[row * 256 + c] = (_Float16)y;
  }
}

// ---------------------------------------------------------------- fused mamba block
__global__ __launch_bounds__(256, 1) void mamba_fused(
    const float* __restrict__ x, const float* __restrict__ ln_g,
    const float* __restrict__ ln_b, const float* __restrict__ conv_w,
    const float* __restrict__ conv_b, const float* __restrict__ dt_proj_b,
    const float* __restrict__ Dvec, const float* __restrict__ proj_b,
    const _Float16* __restrict__ wh, float* __restrict__ out) {
  // 32KB arena: phase 0 = f32 x-tile (TDM dest); phase >=1 = bufB(16K) + zbuf(16K)
  __shared__ __align__(16) char arena[32 * 1024];
  __shared__ __align__(16) _Float16 bufA[ROWS * 256];  // xn, later y (16KB)
  __shared__ __align__(16) float    xdb[ROWS * 48];
  __shared__ __align__(16) _Float16 dtA[ROWS * 32];
  __shared__ float bc_s[ROWS];

  float*    xtile = (float*)arena;                       // [32][256] f32
  _Float16* bufB  = (_Float16*)arena;                    // [32][256] f16 (xc, later t)
  _Float16* zbuf  = (_Float16*)(arena + 16 * 1024);      // [32][256] f16 (silu(z))

  const int tid  = threadIdx.x;
  const int lane = tid & 31;
  const int wv   = tid >> 5;
  const int l16  = lane & 15;
  const int hi8  = (lane >> 4) * 8;
  const long R0  = (long)blockIdx.x * ROWS;

  // ---- Phase -1: TDM async tensor load of the x tile into LDS --------------
#if HAVE_TDM
  if (wv == 0) {
    unsigned long long ga = (unsigned long long)(const void*)(x + R0 * D_MODEL);
    unsigned int lds_off = (unsigned int)(size_t)(void*)xtile;
    u32x4 g0;
    g0[0] = 1u;                                          // count=1, user mode
    g0[1] = lds_off;                                     // lds_addr
    g0[2] = (unsigned int)(ga & 0xffffffffu);            // global_addr[31:0]
    g0[3] = (unsigned int)((ga >> 32) & 0x01ffffffu) | (2u << 30);  // [56:32] | type=2
    i32x8 g1;
    g1[0] = (int)(2u << 16);       // data_size=4B, no mask/flags
    g1[1] = (int)(256u << 16);     // tensor_dim0 = 256 (bits 79:48, low half)
    g1[2] = (int)(32u << 16);      // tensor_dim1 = 32  (bits 111:80, low half)
    g1[3] = (int)(256u << 16);     // tile_dim0 = 256   (bits 127:112)
    g1[4] = 32;                    // tile_dim1 = 32    (bits 143:128)
    g1[5] = 256;                   // tensor_dim0_stride = 256 (bits 207:160, low)
    g1[6] = 0;
    g1[7] = 0;
    i32x4 z4 = {0, 0, 0, 0};
#if defined(__clang_major__) && __clang_major__ >= 23
    i32x8 z8 = {0, 0, 0, 0, 0, 0, 0, 0};
    __builtin_amdgcn_tensor_load_to_lds(g0, g1, z4, z4, z8, 0);
#else
    __builtin_amdgcn_tensor_load_to_lds(g0, g1, z4, z4, 0);
#endif
    __builtin_amdgcn_s_wait_tensorcnt(0);
  }
  __syncthreads();
#endif

  // ---- Phase 0: LayerNorm, f32 -> f16 A tile -------------------------------
  for (int i = 0; i < 4; ++i) {
    const int row = wv * 4 + i;
#if HAVE_TDM
    const float* xr = xtile + row * D_MODEL + lane * 8;
#else
    const float* xr = x + (R0 + row) * D_MODEL + lane * 8;
#endif
    float4 v0 = *(const float4*)xr;
    float4 v1 = *(const float4*)(xr + 4);
    float s = v0.x + v0.y + v0.z + v0.w + v1.x + v1.y + v1.z + v1.w;
    float q = v0.x * v0.x + v0.y * v0.y + v0.z * v0.z + v0.w * v0.w +
              v1.x * v1.x + v1.y * v1.y + v1.z * v1.z + v1.w * v1.w;
#pragma unroll
    for (int off = 16; off > 0; off >>= 1) {
      s += __shfl_xor(s, off, 32);
      q += __shfl_xor(q, off, 32);
    }
    float mu = s * (1.f / D_MODEL);
    float var = q * (1.f / D_MODEL) - mu * mu;
    float rs = rsqrtf(var + EPS);
    float4 g0 = *(const float4*)(ln_g + lane * 8);
    float4 g1 = *(const float4*)(ln_g + lane * 8 + 4);
    float4 b0 = *(const float4*)(ln_b + lane * 8);
    float4 b1 = *(const float4*)(ln_b + lane * 8 + 4);
    h8 o;
    o[0] = (_Float16)((v0.x - mu) * rs * g0.x + b0.x);
    o[1] = (_Float16)((v0.y - mu) * rs * g0.y + b0.y);
    o[2] = (_Float16)((v0.z - mu) * rs * g0.z + b0.z);
    o[3] = (_Float16)((v0.w - mu) * rs * g0.w + b0.w);
    o[4] = (_Float16)((v1.x - mu) * rs * g1.x + b1.x);
    o[5] = (_Float16)((v1.y - mu) * rs * g1.y + b1.y);
    o[6] = (_Float16)((v1.z - mu) * rs * g1.z + b1.z);
    o[7] = (_Float16)((v1.w - mu) * rs * g1.w + b1.w);
    *(h8*)(bufA + row * 256 + lane * 8) = o;
  }
  __syncthreads();

  // ---- Phase 1: GEMM1 xz = xn @ in_proj^T (dual-accumulator), fused SiLU ---
  {
    const int m0 = (wv & 1) * 16;        // row-tile parity fixed per wave
    const int nbase = (wv >> 1) * 8;     // 8 consecutive n-tiles per wave group
    for (int nn = 0; nn < 8; nn += 2) {
      const int n0 = (nbase + nn) * 16;
      f8 acc0 = {}, acc1 = {};
      for (int k0 = 0; k0 < 256; k0 += 32) {
        h16 a  = ld_a(bufA + m0 * 256 + k0, 256, lane);
        h16 b0 = ld_b(wh + OFF_INPROJ, 256, n0,      k0, lane);
        h16 b1 = ld_b(wh + OFF_INPROJ, 256, n0 + 16, k0, lane);
        acc0 = wmma_step(a, b0, acc0);
        acc1 = wmma_step(a, b1, acc1);
      }
      epi_xz(acc0, n0,      m0, l16, hi8, conv_w, conv_b, bufB, zbuf);
      epi_xz(acc1, n0 + 16, m0, l16, hi8, conv_w, conv_b, bufB, zbuf);
    }
  }
  __syncthreads();

  // ---- Phase 2: GEMM2 xdb = xc @ x_proj^T  [32 x 48] -----------------------
  if (wv < 6) {
    const int m0 = (wv & 1) * 16;
    const int n0 = (wv >> 1) * 16;
    f8 acc = {};
    for (int k0 = 0; k0 < 256; k0 += 32) {
      h16 a = ld_a(bufB + m0 * 256 + k0, 256, lane);
      h16 b = ld_b(wh + OFF_XPROJ, 256, n0, k0, lane);
      acc = wmma_step(a, b, acc);
    }
    const int c = n0 + l16;
#pragma unroll
    for (int r = 0; r < 8; ++r) xdb[(m0 + r + hi8) * 48 + c] = acc[r];
  }
  __syncthreads();

  // ---- Phase 3: dt_low (K-padded f16) + bc = <Bs,Cs> -----------------------
  if (tid < ROWS) {
    const int row = tid;
#pragma unroll
    for (int k = 0; k < 32; ++k)
      dtA[row * 32 + k] = (k < 16) ? (_Float16)xdb[row * 48 + k] : (_Float16)0.f;
    float s = 0.f;
#pragma unroll
    for (int j = 0; j < 16; ++j)
      s += xdb[row * 48 + 16 + j] * xdb[row * 48 + 32 + j];
    bc_s[row] = s;
  }
  __syncthreads();

  // ---- Phase 4: dt = softplus(dt_low @ dt_proj^T + b), fused y -------------
  {
    const int m0 = (wv & 1) * 16;
    const int nbase = (wv >> 1) * 4;     // 16 n-tiles / 4 wave groups
    h16 a = ld_a(dtA + m0 * 32, 32, lane);   // invariant across tiles
    for (int nn = 0; nn < 4; nn += 2) {
      const int n0a = (nbase + nn) * 16;
      const int n0b = n0a + 16;
      h16 b0 = ld_b(wh + OFF_DTPROJ, 32, n0a, 0, lane);
      h16 b1 = ld_b(wh + OFF_DTPROJ, 32, n0b, 0, lane);
      f8 zz = {};
      f8 acc0 = wmma_step(a, b0, zz);
      f8 acc1 = wmma_step(a, b1, zz);
      epi_y(acc0, n0a, m0, l16, hi8, dt_proj_b, Dvec, bufB, zbuf, bc_s, bufA);
      epi_y(acc1, n0b, m0, l16, hi8, dt_proj_b, Dvec, bufB, zbuf, bc_s, bufA);
    }
  }
  __syncthreads();

  // ---- Phase 5: GEMM3 t = y @ out_proj^T (dual accumulator) ----------------
  {
    const int m0 = (wv & 1) * 16;
    const int nbase = (wv >> 1) * 4;
    for (int nn = 0; nn < 4; nn += 2) {
      const int n0 = (nbase + nn) * 16;
      f8 acc0 = {}, acc1 = {};
      for (int k0 = 0; k0 < 256; k0 += 32) {
        h16 a  = ld_a(bufA + m0 * 256 + k0, 256, lane);
        h16 b0 = ld_b(wh + OFF_OUTPROJ, 256, n0,      k0, lane);
        h16 b1 = ld_b(wh + OFF_OUTPROJ, 256, n0 + 16, k0, lane);
        acc0 = wmma_step(a, b0, acc0);
        acc1 = wmma_step(a, b1, acc1);
      }
      const int c0 = n0 + l16, c1 = n0 + 16 + l16;
#pragma unroll
      for (int r = 0; r < 8; ++r) {
        int row = m0 + r + hi8;
        bufB[row * 256 + c0] = (_Float16)acc0[r];
        bufB[row * 256 + c1] = (_Float16)acc1[r];
      }
    }
  }
  __syncthreads();

  // ---- Phase 6: GEMM4 out = t @ proj^T + proj_b + residual -----------------
  {
    const int m0 = (wv & 1) * 16;
    const int nbase = (wv >> 1) * 4;
    for (int nn = 0; nn < 4; nn += 2) {
      const int n0 = (nbase + nn) * 16;
      f8 acc0 = {}, acc1 = {};
      for (int k0 = 0; k0 < 256; k0 += 32) {
        h16 a  = ld_a(bufB + m0 * 256 + k0, 256, lane);
        h16 b0 = ld_b(wh + OFF_PROJ, 256, n0,      k0, lane);
        h16 b1 = ld_b(wh + OFF_PROJ, 256, n0 + 16, k0, lane);
        acc0 = wmma_step(a, b0, acc0);
        acc1 = wmma_step(a, b1, acc1);
      }
      const int c0 = n0 + l16, c1 = n0 + 16 + l16;
      float pb0 = proj_b[c0], pb1 = proj_b[c1];
#pragma unroll
      for (int r = 0; r < 8; ++r) {
        long gr = R0 + m0 + r + hi8;
        out[gr * 256 + c0] = acc0[r] + pb0 + x[gr * 256 + c0];
        out[gr * 256 + c1] = acc1[r] + pb1 + x[gr * 256 + c1];
      }
    }
  }
}

// ---------------------------------------------------------------- launcher
extern "C" void kernel_launch(void* const* d_in, const int* in_sizes, int n_in,
                              void* d_out, int out_size, void* d_ws, size_t ws_size,
                              hipStream_t stream) {
  const float* x        = (const float*)d_in[0];
  const float* ln_g     = (const float*)d_in[1];
  const float* ln_b     = (const float*)d_in[2];
  const float* in_proj  = (const float*)d_in[3];
  const float* conv_w   = (const float*)d_in[4];
  const float* conv_b   = (const float*)d_in[5];
  const float* x_proj   = (const float*)d_in[6];
  const float* dt_proj  = (const float*)d_in[7];
  const float* dt_pb    = (const float*)d_in[8];
  // d_in[9] = A_log (unused by the reference computation)
  const float* Dvec     = (const float*)d_in[10];
  const float* out_proj = (const float*)d_in[11];
  const float* proj_w   = (const float*)d_in[12];
  const float* proj_b   = (const float*)d_in[13];

  _Float16* wh = (_Float16*)d_ws;

  prep_weights<<<WS_HALFS / 256, 256, 0, stream>>>(in_proj, x_proj, dt_proj,
                                                   out_proj, proj_w, wh);

  const int batch = in_sizes[0] / D_MODEL;   // 131072
  mamba_fused<<<batch / ROWS, 256, 0, stream>>>(x, ln_g, ln_b, conv_w, conv_b,
                                                dt_pb, Dvec, proj_b, wh,
                                                (float*)d_out);
}
